// MVVNeRFRenderer_20847771254970
// MI455X (gfx1250) — matrix operand order repaired and checked by reading, not compiled, not run on hardware
//
#include <hip/hip_runtime.h>
#include <hip/hip_bf16.h>

typedef __attribute__((ext_vector_type(16))) _Float16 v16h;
typedef __attribute__((ext_vector_type(8)))  _Float16 v8h;
typedef __attribute__((ext_vector_type(8)))  float    v8f;

#define B_    1
#define V_    2
#define R_    512
#define P_    64
#define H_    480
#define W_    640
#define F_    256
#define NCH   259          // 3 (img) + F
#define DIN_  265          // 3 + 3 + 259
#define K1PAD 288          // DIN padded to multiple of 32
#define HID   128
#define NEAR_ 0.7f
#define FAR_  1.5f
#define RAYS_PER_BLOCK 4
#define TILES_PER_BLOCK (RAYS_PER_BLOCK * (P_/16))

// ---------- prep: fp32 weights -> fp16, transposed, K-padded ----------
__global__ void prep_weights(const float* __restrict__ W1, const float* __restrict__ W2,
                             _Float16* __restrict__ w1t, _Float16* __restrict__ w2t) {
  int idx = blockIdx.x * blockDim.x + threadIdx.x;
  const int n1 = HID * K1PAD;
  if (idx < n1) {
    int f = idx / K1PAD, d = idx % K1PAD;
    w1t[idx] = (d < DIN_) ? (_Float16)W1[d * HID + f] : (_Float16)0.f;
  } else {
    int i2 = idx - n1;
    if (i2 < HID * HID) {
      int f = i2 / HID, d = i2 % HID;
      w2t[i2] = (_Float16)W2[d * HID + f];
    }
  }
}

// ---------- fused NeRF render ----------
__global__ __launch_bounds__(256)
void nerf_fused(const float* __restrict__ ro,  const float* __restrict__ rd,
                const float* __restrict__ img, const float* __restrict__ intr,
                const float* __restrict__ extr,const float* __restrict__ feat,
                const _Float16* __restrict__ w1t, const float* __restrict__ b1,
                const _Float16* __restrict__ w2t, const float* __restrict__ b2,
                const float* __restrict__ Wr,  const float* __restrict__ br,
                float* __restrict__ out)
{
  __shared__ _Float16 Xs[16][K1PAD];      // input tile (16 rows x 288)
  __shared__ _Float16 H1s[16][HID];       // layer-1 activations (f16)
  __shared__ float    Hsum[16][HID];      // view-mean accumulator
  __shared__ float    densS[RAYS_PER_BLOCK][P_];
  __shared__ float    chromS[RAYS_PER_BLOCK][P_][3];

  const int tid   = threadIdx.x;
  const int wave  = tid >> 5;
  const int lane  = tid & 31;
  const int lhalf = lane >> 4;            // 0 or 1 (lane group)
  const int l16   = lane & 15;

  // Zero the K-pad columns of Xs once (never rewritten).
  for (int i = tid; i < 16 * (K1PAD - DIN_); i += 256) {
    int rrow = i / (K1PAD - DIN_);
    int kk   = DIN_ + i % (K1PAD - DIN_);
    Xs[rrow][kk] = (_Float16)0.f;
  }
  __syncthreads();

  for (int ti = 0; ti < TILES_PER_BLOCK; ++ti) {
    const int rl = ti / (P_ / 16);
    const int p0 = (ti % (P_ / 16)) * 16;
    const int rg = blockIdx.x * RAYS_PER_BLOCK + rl;

    for (int v = 0; v < V_; ++v) {
      // ---- build X tile: geometry + bilinear gather (16 threads per row) ----
      {
        const int row = tid >> 4;
        const int c0  = tid & 15;
        const int p   = p0 + row;
        const float z = NEAR_ + (FAR_ - NEAR_) * (float)p / (float)(P_ - 1);
        const float ox = ro[rg*3+0], oy = ro[rg*3+1], oz = ro[rg*3+2];
        const float dx = rd[rg*3+0], dy = rd[rg*3+1], dz = rd[rg*3+2];
        const float wxp = ox + z*dx, wyp = oy + z*dy, wzp = oz + z*dz;
        const float* e  = extr + v * 16;
        const float* ki = intr + v * 16;
        float cam[4], dcam[3], pix[3];
        #pragma unroll
        for (int i = 0; i < 4; ++i)
          cam[i] = e[i*4+0]*wxp + e[i*4+1]*wyp + e[i*4+2]*wzp + e[i*4+3];
        #pragma unroll
        for (int i = 0; i < 3; ++i) {
          dcam[i] = e[i*4+0]*dx + e[i*4+1]*dy + e[i*4+2]*dz;
          pix[i]  = ki[i*4+0]*cam[0] + ki[i*4+1]*cam[1] + ki[i*4+2]*cam[2] + ki[i*4+3]*cam[3];
        }
        const float invz = 1.f / (pix[2] + 1e-8f);
        float px = fminf(fmaxf(pix[0]*invz, 0.f), (float)(W_-1));
        float py = fminf(fmaxf(pix[1]*invz, 0.f), (float)(H_-1));
        const int x0 = (int)floorf(px), y0 = (int)floorf(py);
        const int x1 = min(x0+1, W_-1), y1 = min(y0+1, H_-1);
        const float fx = px - (float)x0, fy = py - (float)y0;
        const float w00 = (1.f-fx)*(1.f-fy), w01 = fx*(1.f-fy);
        const float w10 = (1.f-fx)*fy,       w11 = fx*fy;
        const long c00 = (long)(v*H_ + y0)*W_ + x0;
        const long c01 = (long)(v*H_ + y0)*W_ + x1;
        const long c10 = (long)(v*H_ + y1)*W_ + x0;
        const long c11 = (long)(v*H_ + y1)*W_ + x1;
        if (c0 < 3) {
          Xs[row][c0]     = (_Float16)cam[c0];
          Xs[row][3 + c0] = (_Float16)dcam[c0];
        }
        for (int c = c0; c < NCH; c += 16) {
          float s;
          if (c < 3) {
            s = img[c00*3+c]*w00 + img[c01*3+c]*w01 + img[c10*3+c]*w10 + img[c11*3+c]*w11;
            s = 2.f * s - 1.f;
          } else {
            const int cf = c - 3;
            s = feat[c00*F_+cf]*w00 + feat[c01*F_+cf]*w01
              + feat[c10*F_+cf]*w10 + feat[c11*F_+cf]*w11;
          }
          Xs[row][6 + c] = (_Float16)s;
        }
      }
      __syncthreads();

      // ---- GEMM1: (16 x 288) @ (288 x 128), wave w owns columns [16w,16w+16) ----
      {
        v8f acc = {};
        const int ncol = wave * 16 + l16;
        #pragma unroll
        for (int kb = 0; kb < K1PAD / 32; ++kb) {
          // A fragment (contiguous k order, k = 16*lane_group + s): 16 contiguous halves
          const _Float16* arow = &Xs[l16][kb*32 + lhalf*16];
          v8h alo = *(const v8h*)arow;
          v8h ahi = *(const v8h*)(arow + 8);
          v16h a;
          #pragma unroll
          for (int j = 0; j < 8; ++j) { a[j] = alo[j]; a[8+j] = ahi[j]; }
          // B fragment: 16 contiguous halves of transposed weights (L2-resident)
          v16h b = *(const v16h*)(w1t + (long)ncol*K1PAD + kb*32 + lhalf*16);
          acc = __builtin_amdgcn_wmma_f32_16x16x32_f16(false, a, false, b,
                                                       (short)0, acc, false, false);
        }
        const float bias = b1[ncol];
        #pragma unroll
        for (int i = 0; i < 8; ++i) {
          const int m = lhalf * 8 + i;          // C/D layout: M = i + 8*lane_group
          H1s[m][ncol] = (_Float16)fmaxf(acc[i] + bias, 0.f);
        }
      }
      __syncthreads();

      // ---- GEMM2: (16 x 128) @ (128 x 128) + view-mean accumulate ----
      {
        v8f acc = {};
        const int ncol = wave * 16 + l16;
        #pragma unroll
        for (int kb = 0; kb < HID / 32; ++kb) {
          const _Float16* arow = &H1s[l16][kb*32 + lhalf*16];
          v8h alo = *(const v8h*)arow;
          v8h ahi = *(const v8h*)(arow + 8);
          v16h a;
          #pragma unroll
          for (int j = 0; j < 8; ++j) { a[j] = alo[j]; a[8+j] = ahi[j]; }
          v16h b = *(const v16h*)(w2t + (long)ncol*HID + kb*32 + lhalf*16);
          acc = __builtin_amdgcn_wmma_f32_16x16x32_f16(false, a, false, b,
                                                       (short)0, acc, false, false);
        }
        const float bias = b2[ncol];
        #pragma unroll
        for (int i = 0; i < 8; ++i) {
          const int m = lhalf * 8 + i;
          const float h = fmaxf(acc[i] + bias, 0.f) * (1.f / (float)V_);
          if (v == 0) Hsum[m][ncol] = h; else Hsum[m][ncol] += h;
        }
      }
      __syncthreads();
    } // view loop

    // ---- readout: 16 rows x 4 outputs, one (row,o) per thread ----
    if (tid < 64) {
      const int row = tid >> 2, o = tid & 3;
      float acc = br[o];
      for (int d = 0; d < HID; ++d) acc += Hsum[row][d] * Wr[d*4 + o];
      const int p = p0 + row;
      if (o == 3) densS[rl][p]     = fmaxf(acc, 0.f);
      else        chromS[rl][p][o] = 1.f / (1.f + __expf(-acc));
    }
    __syncthreads();
  } // tile loop

  // ---- volumetric render: exclusive-cumprod scan, one thread per ray ----
  if (tid < RAYS_PER_BLOCK) {
    const int rg = blockIdx.x * RAYS_PER_BLOCK + tid;
    const float dist = (FAR_ - NEAR_) / (float)(P_ - 1);
    float T = 1.f, r0 = 0.f, r1 = 0.f, r2 = 0.f, dep = 0.f;
    for (int p = 0; p < P_; ++p) {
      const float z     = NEAR_ + dist * (float)p;
      const float alpha = 1.f - __expf(-densS[tid][p] * dist);
      const float w     = alpha * T;
      r0  += w * chromS[tid][p][0];
      r1  += w * chromS[tid][p][1];
      r2  += w * chromS[tid][p][2];
      dep += w * z;
      T   *= (1.f - alpha + 1e-10f);
    }
    out[rg*3 + 0] = r0;
    out[rg*3 + 1] = r1;
    out[rg*3 + 2] = r2;
    out[R_*3 + rg] = dep;          // depth follows rgb in flat output
  }
}

extern "C" void kernel_launch(void* const* d_in, const int* in_sizes, int n_in,
                              void* d_out, int out_size, void* d_ws, size_t ws_size,
                              hipStream_t stream) {
  const float* ro   = (const float*)d_in[0];
  const float* rd   = (const float*)d_in[1];
  const float* img  = (const float*)d_in[2];
  const float* intr = (const float*)d_in[3];
  const float* extr = (const float*)d_in[4];
  const float* feat = (const float*)d_in[5];
  const float* W1   = (const float*)d_in[6];
  const float* b1   = (const float*)d_in[7];
  const float* W2   = (const float*)d_in[8];
  const float* b2   = (const float*)d_in[9];
  const float* Wr   = (const float*)d_in[10];
  const float* br   = (const float*)d_in[11];

  _Float16* w1t = (_Float16*)d_ws;
  _Float16* w2t = w1t + HID * K1PAD;

  const int total = HID * K1PAD + HID * HID;
  prep_weights<<<(total + 255) / 256, 256, 0, stream>>>(W1, W2, w1t, w2t);
  nerf_fused<<<R_ / RAYS_PER_BLOCK, 256, 0, stream>>>(ro, rd, img, intr, extr, feat,
                                                      w1t, b1, w2t, b2, Wr, br,
                                                      (float*)d_out);
}
